// BallMSA_42880953483553
// MI455X (gfx1250) — compile-verified, matrix operand order checked by reading, not run on hardware
//
#include <hip/hip_runtime.h>
#include <hip/hip_bf16.h>
#include <math.h>

// BallMSA fully-fused kernel for gfx1250 (MI455X).
//   k_convert : weights fp32 -> bf16, pre-TRANSPOSED ([col][k]) + QKV column
//               permute (H,E,3)->(3,H,E) so WMMA B-fragments load contiguously.
//   k_ballmsa_full : one workgroup per 2 balls (64 rows), 512 threads = 16 waves.
//               phase0: pos-embed x' -> LDS (bf16)
//               phase1: QKV GEMM 64x256 @ 256x768, barrier-free K-loop
//                       (A from resident LDS, B streamed from L2 weights)
//               phase2: per-(ball,head) 32x32 attention, wave-shuffle softmax
//               phase3: proj GEMM 64x256 @ 256x256 -> fp32 out
// HBM traffic ~540MB (~23us @23.3TB/s) vs ~1.9GB for an unfused pipeline;
// ~155 GFLOP of bf16 WMMA (~30us) -> roughly balanced at the roofline.
// Only possible because CDNA5 gives 320KB LDS per WGP (we use ~201KB).

#define NHEAD 8
#define MSZ 32
#define NBALL 8192
#define NPT (NBALL * MSZ)
#define DIMC 256
#define BPW 2                     // balls per workgroup
#define ROWS (BPW * MSZ)          // 64
#define NTHREADS 512              // 16 waves

typedef __bf16 bf16_t;
typedef __bf16 v16bf __attribute__((ext_vector_type(16)));
typedef float  v8f   __attribute__((ext_vector_type(8)));

union AF { v16bf v; uint4 q[2]; };

// 16x32 bf16 WMMA A/B fragment from a row-major tile.
// lane<16: row=lane, K in {0..7}(v0-3) + {16..23}(v4-7); lane>=16: row=lane-16,
// K in {8..15} + {24..31}.
__device__ __forceinline__ v16bf load_frag(const bf16_t* base, int row, int pitch, int lane) {
    const int half = lane >> 4;
    const int mr   = lane & 15;
    const bf16_t* p = base + (size_t)(row + mr) * pitch;
    AF f;
    f.q[0] = *(const uint4*)(p + (half << 3));
    f.q[1] = *(const uint4*)(p + 16 + (half << 3));
    return f.v;
}

__device__ __forceinline__ v8f wmma_bf16(v16bf a, v16bf b, v8f c) {
    return __builtin_amdgcn_wmma_f32_16x16x32_bf16(false, a, false, b, (short)0, c, false, false);
}

// LDS partition byte offsets (dynamic shared memory, 205824 B total)
#define SM_POS   0        // 64*3 f32        (768 -> pad 1024)
#define SM_DIST  1024     // 2*32*32 f32     (8192)
#define SM_X     9216     // 64*256 bf16     (32768)  -- overlaid by attnS
#define SM_ATTN  9216     // 16*1024 bf16    (32768)
#define SM_QKV   41984    // 64*768 bf16     (98304)
#define SM_VT    140288   // 16*1024 bf16    (32768)
#define SM_O     173056   // 64*256 bf16     (32768)
#define SM_TOTAL 205824

// ---------------------------------------------------------------- k_convert
__global__ __launch_bounds__(256)
void k_convert(const float* __restrict__ wqkv, const float* __restrict__ bqkv,
               const float* __restrict__ wproj,
               bf16_t* __restrict__ WqkvT, float* __restrict__ bqkvP,
               bf16_t* __restrict__ WprojT) {
    int idx = blockIdx.x * 256 + threadIdx.x;
    if (idx < 768 * 256) {
        // WqkvT[c2][k], c2 = mat*256 + h*32 + e  <-  wqkv[k][h*96 + e*3 + mat]
        int c2 = idx >> 8, k = idx & 255;
        int mat = c2 >> 8, he = c2 & 255, h = he >> 5, e = he & 31;
        WqkvT[idx] = (bf16_t)wqkv[k * 768 + h * 96 + e * 3 + mat];
    } else if (idx < 768 * 256 + 768) {
        int c2 = idx - 768 * 256;
        int mat = c2 >> 8, he = c2 & 255, h = he >> 5, e = he & 31;
        bqkvP[c2] = bqkv[h * 96 + e * 3 + mat];
    } else if (idx < 768 * 256 + 768 + 256 * 256) {
        int i = idx - (768 * 256 + 768);
        int n = i >> 8, k = i & 255;
        WprojT[i] = (bf16_t)wproj[k * 256 + n];
    }
}

// ---------------------------------------------------------------- k_ballmsa
__global__ __launch_bounds__(NTHREADS)
void k_ballmsa_full(const float* __restrict__ x, const float* __restrict__ pos,
                    const float* __restrict__ wpe, const float* __restrict__ bpe,
                    const bf16_t* __restrict__ WqkvT, const float* __restrict__ bqkvP,
                    const bf16_t* __restrict__ WprojT, const float* __restrict__ bproj,
                    const float* __restrict__ sigma, float* __restrict__ out) {
    extern __shared__ char smem[];
    float*  posL  = (float*)(smem + SM_POS);
    float*  distL = (float*)(smem + SM_DIST);
    bf16_t* xS    = (bf16_t*)(smem + SM_X);
    bf16_t* attnS = (bf16_t*)(smem + SM_ATTN);   // overlays xS (dead after phase1)
    bf16_t* qkvS  = (bf16_t*)(smem + SM_QKV);
    bf16_t* vtS   = (bf16_t*)(smem + SM_VT);
    bf16_t* oS    = (bf16_t*)(smem + SM_O);

    const int tid  = threadIdx.x;
    const int lane = tid & 31, wave = tid >> 5;
    const int half = lane >> 4, nl = lane & 15;
    const size_t rowBase = (size_t)blockIdx.x * ROWS;

    // ---------------- phase 0: pos, dist, x' = x + rel@w_pe + b_pe (bf16)
    if (tid < ROWS) {
        size_t g = rowBase + tid;
        posL[tid * 3 + 0] = pos[g * 3 + 0];
        posL[tid * 3 + 1] = pos[g * 3 + 1];
        posL[tid * 3 + 2] = pos[g * 3 + 2];
    }
    __syncthreads();
    for (int t = tid; t < BPW * MSZ * MSZ; t += NTHREADS) {
        int b = t >> 10, ij = t & 1023, i = ij >> 5, j = ij & 31;
        float dx = posL[(b*32+i)*3+0] - posL[(b*32+j)*3+0];
        float dy = posL[(b*32+i)*3+1] - posL[(b*32+j)*3+1];
        float dz = posL[(b*32+i)*3+2] - posL[(b*32+j)*3+2];
        distL[t] = sqrtf(fmaxf(dx*dx + dy*dy + dz*dz, 1e-12f));
    }
    {
        const int r = tid >> 3, c0 = (tid & 7) * 32;   // 8 threads per row
        const int b = r >> 5;
        float cx = 0.f, cy = 0.f, cz = 0.f;
        #pragma unroll
        for (int i = 0; i < 32; i++) {
            cx += posL[(b*32+i)*3+0]; cy += posL[(b*32+i)*3+1]; cz += posL[(b*32+i)*3+2];
        }
        cx *= 0.03125f; cy *= 0.03125f; cz *= 0.03125f;
        const float rx = posL[r*3+0] - cx, ry = posL[r*3+1] - cy, rz = posL[r*3+2] - cz;
        const float* xr = x + (rowBase + r) * DIMC;
        #pragma unroll 8
        for (int c = c0; c < c0 + 32; c++) {
            float v = xr[c] + rx * wpe[c] + ry * wpe[256 + c] + rz * wpe[512 + c] + bpe[c];
            xS[r * DIMC + c] = (bf16_t)v;
        }
    }
    __syncthreads();

    // ---------------- phase 1: QKV GEMM (barrier-free K loop)
    // wave w -> cols [w*48, w*48+48), all 64 rows: 4x3 WMMA accumulators
    {
        const int cb = wave * 48;
        v8f acc[4][3] = {};
        for (int k0 = 0; k0 < 256; k0 += 32) {
            v16bf b0 = load_frag(WqkvT + k0, cb,      256, lane);
            v16bf b1 = load_frag(WqkvT + k0, cb + 16, 256, lane);
            v16bf b2 = load_frag(WqkvT + k0, cb + 32, 256, lane);
            #pragma unroll
            for (int rt = 0; rt < 4; rt++) {
                v16bf a = load_frag(xS + k0, rt * 16, 256, lane);
                acc[rt][0] = wmma_bf16(a, b0, acc[rt][0]);
                acc[rt][1] = wmma_bf16(a, b1, acc[rt][1]);
                acc[rt][2] = wmma_bf16(a, b2, acc[rt][2]);
            }
        }
        #pragma unroll
        for (int j = 0; j < 3; j++) {
            const int col = cb + j * 16 + nl;
            const float bv = bqkvP[col];
            #pragma unroll
            for (int rt = 0; rt < 4; rt++)
                #pragma unroll
                for (int r = 0; r < 8; r++) {
                    int row = rt * 16 + r + 8 * half;
                    qkvS[row * 768 + col] = (bf16_t)(acc[rt][j][r] + bv);
                }
        }
    }
    __syncthreads();

    // ---------------- phase 2: attention; wave = (ball b2, head h)
    {
        const int b2 = wave >> 3, h = wave & 7;
        const bf16_t* qS = qkvS + b2 * 32 * 768;
        v16bf aq0 = load_frag(qS + h * 32,        0, 768, lane);
        v16bf aq1 = load_frag(qS + h * 32,       16, 768, lane);
        v16bf bk0 = load_frag(qS + 256 + h * 32,  0, 768, lane);
        v16bf bk1 = load_frag(qS + 256 + h * 32, 16, 768, lane);
        v8f s[2][2] = {};
        s[0][0] = wmma_bf16(aq0, bk0, s[0][0]);
        s[0][1] = wmma_bf16(aq0, bk1, s[0][1]);
        s[1][0] = wmma_bf16(aq1, bk0, s[1][0]);
        s[1][1] = wmma_bf16(aq1, bk1, s[1][1]);

        // stage V^T while scores are in flight
        bf16_t* vt = vtS + wave * 1024;
        {
            const bf16_t* vr = qS + (size_t)lane * 768 + 512 + h * 32;
            #pragma unroll
            for (int e = 0; e < 32; e++) vt[e * 32 + lane] = vr[e];
        }

        const float sg    = sigma[h];
        const float scale = 0.17677669529663687f;  // 1/sqrt(32)
        const float* dL   = distL + b2 * 1024;
        bf16_t* at = attnS + wave * 1024;
        #pragma unroll
        for (int ti = 0; ti < 2; ti++)
            #pragma unroll
            for (int r = 0; r < 8; r++) {
                const int m = ti * 16 + r + 8 * half;
                float v0 = s[ti][0][r] * scale + sg * dL[m * 32 + nl];
                float v1 = s[ti][1][r] * scale + sg * dL[m * 32 + 16 + nl];
                float mx = fmaxf(v0, v1);
                #pragma unroll
                for (int off = 8; off >= 1; off >>= 1) mx = fmaxf(mx, __shfl_xor(mx, off, 32));
                float e0 = __expf(v0 - mx), e1 = __expf(v1 - mx);
                float sum = e0 + e1;
                #pragma unroll
                for (int off = 8; off >= 1; off >>= 1) sum += __shfl_xor(sum, off, 32);
                float invs = 1.0f / sum;
                at[m * 32 + nl]      = (bf16_t)(e0 * invs);
                at[m * 32 + 16 + nl] = (bf16_t)(e1 * invs);
            }
        __syncthreads();

        v16bf aa0 = load_frag(at,  0, 32, lane);
        v16bf aa1 = load_frag(at, 16, 32, lane);
        v16bf bv0 = load_frag(vt,  0, 32, lane);
        v16bf bv1 = load_frag(vt, 16, 32, lane);
        v8f o[2][2] = {};
        o[0][0] = wmma_bf16(aa0, bv0, o[0][0]);
        o[0][1] = wmma_bf16(aa0, bv1, o[0][1]);
        o[1][0] = wmma_bf16(aa1, bv0, o[1][0]);
        o[1][1] = wmma_bf16(aa1, bv1, o[1][1]);
        #pragma unroll
        for (int ti = 0; ti < 2; ti++)
            #pragma unroll
            for (int tj = 0; tj < 2; tj++)
                #pragma unroll
                for (int r = 0; r < 8; r++) {
                    int row = b2 * 32 + ti * 16 + r + 8 * half;
                    int col = h * 32 + tj * 16 + nl;
                    oS[row * DIMC + col] = (bf16_t)o[ti][tj][r];
                }
    }
    __syncthreads();

    // ---------------- phase 3: output projection; wave w -> cols [w*16, w*16+16)
    {
        const int cp = wave * 16;
        v8f acc2[4] = {};
        for (int k0 = 0; k0 < 256; k0 += 32) {
            v16bf b = load_frag(WprojT + k0, cp, 256, lane);
            #pragma unroll
            for (int rt = 0; rt < 4; rt++) {
                v16bf a = load_frag(oS + k0, rt * 16, 256, lane);
                acc2[rt] = wmma_bf16(a, b, acc2[rt]);
            }
        }
        const int col = cp + nl;
        const float bv = bproj[col];
        #pragma unroll
        for (int rt = 0; rt < 4; rt++)
            #pragma unroll
            for (int r = 0; r < 8; r++) {
                size_t row = rowBase + rt * 16 + r + 8 * half;
                out[row * DIMC + col] = acc2[rt][r] + bv;
            }
    }
}

// ---------------------------------------------------------------- launch
extern "C" void kernel_launch(void* const* d_in, const int* in_sizes, int n_in,
                              void* d_out, int out_size, void* d_ws, size_t ws_size,
                              hipStream_t stream) {
    const float* x     = (const float*)d_in[0];
    const float* pos   = (const float*)d_in[1];
    const float* wqkv  = (const float*)d_in[2];
    const float* bqkv  = (const float*)d_in[3];
    const float* wpe   = (const float*)d_in[4];
    const float* bpe   = (const float*)d_in[5];
    const float* wproj = (const float*)d_in[6];
    const float* bproj = (const float*)d_in[7];
    const float* sigma = (const float*)d_in[8];
    float* out = (float*)d_out;
    (void)in_sizes; (void)n_in; (void)out_size; (void)ws_size;

    char* ws = (char*)d_ws;
    bf16_t* WqkvT  = (bf16_t*)(ws);                       // 768*256*2 = 393216
    bf16_t* WprojT = (bf16_t*)(ws + 393216);              // 256*256*2 = 131072
    float*  bqkvP  = (float*)(ws + 393216 + 131072);      // 768*4

    // Idempotent + deterministic (called every time; not a stream op, safe
    // under graph capture): allow >64KB dynamic LDS for the fused kernel.
    hipFuncSetAttribute((const void*)k_ballmsa_full,
                        hipFuncAttributeMaxDynamicSharedMemorySize, SM_TOTAL);

    k_convert<<<1027, 256, 0, stream>>>(wqkv, bqkv, wproj, WqkvT, bqkvP, WprojT);
    k_ballmsa_full<<<NBALL / BPW, NTHREADS, SM_TOTAL, stream>>>(
        x, pos, wpe, bpe, WqkvT, bqkvP, WprojT, bproj, sigma, out);
}
